// PaiConvISO_10050223472787
// MI455X (gfx1250) — matrix operand order repaired and checked by reading, not compiled, not run on hardware
//
#include <hip/hip_runtime.h>
#include <hip/hip_bf16.h>
#include <stdint.h>

// Problem constants (from reference setup_inputs)
#define B_    8
#define C_    64
#define N_    4096
#define K_    20
#define KS_   20
#define OUT_  64
#define BN_   (B_ * N_)     // 32768 points
#define KPAD_ 32            // K and KS padded to 32 for WMMA
#define WAVES_ 4            // waves (points) per workgroup

// LDS row strides padded to odd multiples of 4B -> conflict-free banking
// (17 and 34 words are coprime-ish with the 64-bank layout: 34*l mod 64 and
//  17*s mod 64 are distinct across a wave's lanes).
#define FE_STRIDE 34        // feats  [64][34] f16, rows 68 B
#define PM_STRIDE 34        // perm^T [32][34] f16
#define F2_STRIDE 68        // F2^T   [32][68] f16, rows 136 B
#define W_STRIDE  68        // conv_w [64][68] f16

typedef __attribute__((ext_vector_type(16))) _Float16 v16h;
typedef __attribute__((ext_vector_type(8)))  float    v8f;

union H2 { uint32_t u; _Float16 h[2]; };

// ---- WMMA helper (CDNA5 V_WMMA_F32_16X16X32_F16, f32 accumulate) ----
__device__ __forceinline__ v8f wmma16(v16h a, v16h b, v8f c) {
  return __builtin_amdgcn_wmma_f32_16x16x32_f16(
      /*neg_a=*/false, a, /*neg_b=*/false, b,
      /*c_mod=*/(short)0, c, /*reuse_a=*/false, /*reuse_b=*/false);
}

// A-matrix fragment, 16x32 f16 tile from row-major LDS [M][rowlen].
// ISA layout: lanes 0-15 hold row M=lane with K = {0..7,16..23};
// lanes 16-31 hold same rows with K = {8..15,24..31}; 2 K per VGPR.
__device__ __forceinline__ v16h load_a16(const _Float16* base, int rowlen,
                                         int lane, int m0, int k0) {
  const int row = m0 + (lane & 15);
  const int kb  = k0 + ((lane & 16) ? 8 : 0);
  const uint32_t* rp = (const uint32_t*)(base + row * rowlen);
  v16h a;
#pragma unroll
  for (int j = 0; j < 8; ++j) {
    const int k = kb + ((j < 4) ? (2 * j) : (8 + 2 * j)); // pairs, +16 jump
    H2 w; w.u = rp[k >> 1];
    a[2 * j] = w.h[0]; a[2 * j + 1] = w.h[1];
  }
  return a;
}

// B-matrix fragment, 32x16 f16 tile from LDS stored transposed as [N][rowlen(K)].
// Layout: lane n = lane&15; lanes 0-15 carry K={0..7,16..23}, lanes 16-31 K={8..15,24..31}.
__device__ __forceinline__ v16h load_b16(const _Float16* base, int rowlen,
                                         int lane, int n0, int k0) {
  const int row = n0 + (lane & 15);
  const int kb  = k0 + ((lane & 16) ? 8 : 0);
  const uint32_t* rp = (const uint32_t*)(base + row * rowlen + kb);
  v16h b;
#pragma unroll
  for (int j = 0; j < 8; ++j) {
    H2 w; w.u = rp[(j < 4) ? j : (j + 4)];
    b[2 * j] = w.h[0]; b[2 * j + 1] = w.h[1];
  }
  return b;
}

// Store a 16x16 f32 D tile into LDS as f16, transposed layout [n][F2_STRIDE]
// so it feeds matmul-2 as a B operand. D layout: VGPR r -> m = r + (lane&16?8:0),
// n = lane&15.
__device__ __forceinline__ void store_d_as_b(v8f d, _Float16* base,
                                             int lane, int m0, int n0) {
  const int s  = n0 + (lane & 15);
  const int cb = m0 + ((lane & 16) ? 8 : 0);
  uint32_t* rp = (uint32_t*)(base + s * F2_STRIDE + cb);
#pragma unroll
  for (int j = 0; j < 4; ++j) {
    H2 w;
    w.h[0] = (_Float16)d[2 * j];
    w.h[1] = (_Float16)d[2 * j + 1];
    rp[j] = w.u;
  }
}

// ---- Kernel 0: zero global stats (sum, sumsq per out-channel) ----
__global__ void k_zero_stats(float* g) {
  if (threadIdx.x < 2 * OUT_) g[threadIdx.x] = 0.f;
}

// ---- Kernel 1: [B,C,N] f32 -> [BN,64] f16 gather table with the group
// channel-shuffle pre-applied (c' = (c%16)*4 + c/16). Reads fully coalesced,
// non-temporal (stream-once); writes stay regular so the 4 MB table lands in L2.
__global__ void k_prep(const float* __restrict__ feat, _Float16* __restrict__ flat) {
  const int gid = blockIdx.x * 256 + threadIdx.x;   // over B*C*N = 2M
  const int n = gid & (N_ - 1);
  const int c = (gid >> 12) & (C_ - 1);
  const int b = gid >> 18;
  const int cp = (c & 15) * 4 + (c >> 4);
  const float v = __builtin_nontemporal_load(feat + gid);
  flat[((long long)b * N_ + n) * C_ + cp] = (_Float16)v;
}

// ---- Kernel 2: main fused gather + per-point soft-perm GEMM + conv1x1 + max ----
__global__ __launch_bounds__(128) void k_main(
    const _Float16* __restrict__ flat, const int* __restrict__ neigh,
    const float* __restrict__ perm, const float* __restrict__ conv_w,
    const float* __restrict__ conv_b, float* __restrict__ out,
    float* __restrict__ gStats) {
  __shared__ __attribute__((aligned(16))) _Float16 sW[OUT_ * W_STRIDE];          // 8.5 KB
  __shared__ __attribute__((aligned(16))) _Float16 sFe[WAVES_][C_ * FE_STRIDE];  // 17 KB
  __shared__ __attribute__((aligned(16))) _Float16 sPm[WAVES_][KPAD_ * PM_STRIDE]; // 8.5 KB
  __shared__ __attribute__((aligned(16))) _Float16 sF2[WAVES_][KPAD_ * F2_STRIDE]; // 17 KB
  __shared__ float sStats[2 * OUT_];

  const int tid = threadIdx.x;
  if (tid < 2 * OUT_) sStats[tid] = 0.f;
  // shared f16 copy of conv_w, [o][W_STRIDE] (channel order matches shuffled feats)
#pragma unroll
  for (int i = tid; i < OUT_ * C_; i += 128) {
    sW[(i >> 6) * W_STRIDE + (i & 63)] = (_Float16)conv_w[i];
  }
  __syncthreads();

  const int wv   = tid >> 5;
  const int lane = tid & 31;
  const int p = blockIdx.x * WAVES_ + wv;   // point id, exact coverage
  const int b = p >> 12;
  const int n = p & (N_ - 1);

  _Float16* fe = sFe[wv];
  _Float16* pm = sPm[wv];
  _Float16* f2 = sF2[wv];
  const float* pp = perm + (long long)p * (K_ * KS_);

  // Pull this point's 1.6 KB soft-perm block toward the caches while the wave
  // is busy with the neighbor gather (gfx1250 global_prefetch_b8).
  if (lane < 13) __builtin_prefetch((const char*)pp + lane * 128, 0, 0);

  // zero-pad feats [64][FE_STRIDE] and perm [32][PM_STRIDE]
  {
    const float4 z4 = {0.f, 0.f, 0.f, 0.f};
    float4* zf = (float4*)fe;                       // 4352 B = 272 float4
    for (int i = lane; i < (C_ * FE_STRIDE) / 8; i += 32) zf[i] = z4;
    float4* zp = (float4*)pm;                       // 2176 B = 136 float4
    for (int i = lane; i < (KPAD_ * PM_STRIDE) / 8; i += 32) zp[i] = z4;
  }

  // neighbor gather: for each k, the wave pulls one 64-ch f16 row (L2-resident
  // 4 MB table) and scatters it as column k of the A matrix [c][k].
  // With FE_STRIDE=34 the per-lane b16 scatters are bank-conflict free.
  {
    const int* ni = neigh + (long long)p * K_;
#pragma unroll 4
    for (int k = 0; k < K_; ++k) {
      const long long idx = (long long)ni[k];
      H2 w; w.u = *(const uint32_t*)(flat + idx * (long long)C_ + 2 * lane);
      fe[(2 * lane)     * FE_STRIDE + k] = w.h[0];
      fe[(2 * lane + 1) * FE_STRIDE + k] = w.h[1];
    }
  }
  // soft-permutation matrix -> f16 LDS, transposed to [s][k] (B-operand layout),
  // streamed non-temporally (read-once 52 MB input, keep L2 for the gather table)
  {
    for (int i = lane; i < K_ * KS_; i += 32) {
      const int k = i / KS_;
      const int s = i - k * KS_;
      pm[s * PM_STRIDE + k] = (_Float16)__builtin_nontemporal_load(pp + i);
    }
  }

  // matmul 1: F2[c][s] = feats[c][k] * perm[k][s]   (64x32x32) -> 8 WMMAs
#pragma unroll
  for (int tn = 0; tn < 2; ++tn) {
    const v16h bm = load_b16(pm, PM_STRIDE, lane, tn * 16, 0);
#pragma unroll
    for (int tm = 0; tm < 4; ++tm) {
      const v16h am = load_a16(fe, FE_STRIDE, lane, tm * 16, 0);
      v8f c = {};
      c = wmma16(am, bm, c);
      store_d_as_b(c, f2, lane, tm * 16, tn * 16);   // f16, [s][c]
    }
  }

  // matmul 2: O[o][s] = conv_w[o][c] * F2[c][s]   (64x32x64) -> 16 WMMAs
  v8f acc[4][2];
  {
    const v8f z = {};
#pragma unroll
    for (int tm = 0; tm < 4; ++tm) { acc[tm][0] = z; acc[tm][1] = z; }
  }
#pragma unroll
  for (int kk = 0; kk < 2; ++kk) {
#pragma unroll
    for (int tn = 0; tn < 2; ++tn) {
      const v16h bm = load_b16(f2, F2_STRIDE, lane, tn * 16, kk * 32);
#pragma unroll
      for (int tm = 0; tm < 4; ++tm) {
        const v16h am = load_a16(sW, W_STRIDE, lane, tm * 16, kk * 32);
        acc[tm][tn] = wmma16(am, bm, acc[tm][tn]);
      }
    }
  }

  // max over the 20 *valid* kernel columns (mask the zero-pad: padded columns
  // would otherwise inject conv_b into the max), + bias, + BN stats
  const int sl = lane & 15;
#pragma unroll
  for (int tm = 0; tm < 4; ++tm) {
#pragma unroll
    for (int r = 0; r < 8; ++r) {
      float m = acc[tm][0][r];                        // s = 0..15
      if (sl < 4) m = fmaxf(m, acc[tm][1][r]);        // s = 16..19 only
#pragma unroll
      for (int d = 1; d < 16; d <<= 1)                // butterfly over 16 lanes
        m = fmaxf(m, __shfl_xor(m, d, 32));
      if (sl == 0) {
        const int o = tm * 16 + ((lane & 16) ? 8 : 0) + r;
        const float val = m + conv_b[o];
        out[(((long long)b * OUT_ + o) << 12) + n] = val;
        atomicAdd(&sStats[o], val);
        atomicAdd(&sStats[OUT_ + o], val * val);
      }
    }
  }

  __syncthreads();
  if (tid < 2 * OUT_) atomicAdd(&gStats[tid], sStats[tid]);
}

// ---- Kernel 3: BatchNorm (training-mode batch statistics) in place ----
__global__ void k_bn(float* __restrict__ out, const float* __restrict__ gStats,
                     const float* __restrict__ bn_w, const float* __restrict__ bn_b) {
  const int gid = blockIdx.x * 256 + threadIdx.x;    // over B*OUT*N = 2M
  const int o = (gid >> 12) & (OUT_ - 1);
  const float inv_cnt = 1.f / (float)BN_;
  const float mean = gStats[o] * inv_cnt;
  const float var  = gStats[OUT_ + o] * inv_cnt - mean * mean;
  const float inv  = rsqrtf(var + 1e-5f);
  const float v = (out[gid] - mean) * inv * bn_w[o] + bn_b[o];
  __builtin_nontemporal_store(v, out + gid);
}

extern "C" void kernel_launch(void* const* d_in, const int* in_sizes, int n_in,
                              void* d_out, int out_size, void* d_ws, size_t ws_size,
                              hipStream_t stream) {
  (void)in_sizes; (void)n_in; (void)out_size; (void)ws_size;
  const float* feature = (const float*)d_in[0];
  const int*   neigh   = (const int*)d_in[1];
  const float* permat  = (const float*)d_in[2];
  const float* conv_w  = (const float*)d_in[3];
  const float* conv_b  = (const float*)d_in[4];
  const float* bn_w    = (const float*)d_in[5];
  const float* bn_b    = (const float*)d_in[6];
  float* out = (float*)d_out;

  // workspace: [BN*64] f16 shuffled gather table (4 MB), then 128 f32 stats
  _Float16* flat = (_Float16*)d_ws;
  float* gStats = (float*)((char*)d_ws + (size_t)BN_ * C_ * sizeof(_Float16));

  k_zero_stats<<<1, 128, 0, stream>>>(gStats);
  k_prep<<<(B_ * C_ * N_) / 256, 256, 0, stream>>>(feature, flat);
  k_main<<<BN_ / WAVES_, 128, 0, stream>>>(flat, neigh, permat, conv_w, conv_b,
                                           out, gStats);
  k_bn<<<(B_ * OUT_ * N_) / 256, 256, 0, stream>>>(out, gStats, bn_w, bn_b);
}
